// Model_83141976916211
// MI455X (gfx1250) — compile-verified, matrix operand order checked by reading, not run on hardware
//
#include <hip/hip_runtime.h>
#include <hip/hip_bf16.h>
#include <stdint.h>

// ---------------------------------------------------------------------------
// 6-layer bidirectional GRU for MI455X (gfx1250, wave32, WMMA + TDM).
//  - bf16 operands + f32 accumulation via V_WMMA_F32_16X16X32_BF16
//  - Tensor Data Mover (tensor_load_to_lds) stages all WMMA tiles into LDS;
//    GEMM uses double-buffered tiles with in-order TENSORcnt pipelining
//    (s_wait_tensorcnt 2) so DMA overlaps WMMA math
//  - recurrence: cluster of 8 WGs per direction, W_hh resident in LDS,
//    cluster barrier (s_barrier_signal/wait -3) between time steps
// ---------------------------------------------------------------------------

typedef __attribute__((ext_vector_type(16))) __bf16 v16bf;
typedef __attribute__((ext_vector_type(8)))  __bf16 v8bf;
typedef __attribute__((ext_vector_type(8)))  float  v8f;
typedef __attribute__((ext_vector_type(4)))  unsigned int u32x4;
typedef __attribute__((ext_vector_type(8)))  int i32x8;
typedef __attribute__((ext_vector_type(4)))  int i32x4;

static constexpr int Tt = 512;   // time steps
static constexpr int Bb = 128;   // batch
static constexpr int Hh = 256;   // hidden
static constexpr int G3 = 768;   // 3*H gates
static constexpr int Mrows = Tt * Bb;   // 65536

#if defined(__has_builtin)
#if __has_builtin(__builtin_amdgcn_tensor_load_to_lds)
#define HAS_TDM 1
#endif
#endif
#ifndef HAS_TDM
#define HAS_TDM 0
#endif

__device__ __forceinline__ v16bf cat16(v8bf a, v8bf b) {
    return __builtin_shufflevector(a, b, 0,1,2,3,4,5,6,7,8,9,10,11,12,13,14,15);
}

__device__ __forceinline__ v8f wmma_bf16(v16bf a, v16bf b, v8f c) {
    return __builtin_amdgcn_wmma_f32_16x16x32_bf16(false, a, false, b, (short)0, c, false, false);
}

__device__ __forceinline__ float sigmoidf(float x) {
    return 1.0f / (1.0f + __expf(-x));
}

#if HAS_TDM
// Issue one 2-D TDM tile load: global (rows x width, row stride `stride`,
// all in elements of size 2^dsz_log2 bytes) -> LDS at byte address lds_addr,
// rows stored contiguously. D# packing per CDNA5 ISA.
// tensor_dim == tile_dim here (tile fully in-bounds, no OOB clamping needed).
__device__ __forceinline__ void tdm_load_2d(unsigned lds_addr, const void* gptr,
                                            unsigned width, unsigned rows,
                                            unsigned stride, unsigned dsz_log2) {
    unsigned long long ga = (unsigned long long)(uintptr_t)gptr;
    u32x4 g0;
    g0[0] = 1u;                                        // count=1 (valid), user mode
    g0[1] = lds_addr;                                  // LDS byte address
    g0[2] = (unsigned)ga;                              // global_addr[31:0]
    g0[3] = (unsigned)((ga >> 32) & 0x01ffffffu)       // global_addr[56:32]
          | (2u << 30);                                // type=2 ("image")
    i32x8 g1;
    g1[0] = (int)(dsz_log2 << 16);                     // wg_mask=0, data_size
    g1[1] = (int)((width & 0xffffu) << 16);            // tensor_dim0[15:0]
    g1[2] = (int)(((width >> 16) & 0xffffu)            // tensor_dim0[31:16]
          | ((rows & 0xffffu) << 16));                 // tensor_dim1[15:0]
    g1[3] = (int)(((rows >> 16) & 0xffffu)             // tensor_dim1[31:16]
          | ((width & 0xffffu) << 16));                // tile_dim0
    g1[4] = (int)(rows & 0xffffu);                     // tile_dim1 (tile_dim2=0)
    g1[5] = (int)stride;                               // tensor_dim0_stride[31:0]
    g1[6] = 0;                                         // stride0[47:32], stride1 lo
    g1[7] = 0;
    i32x4 gz = {0, 0, 0, 0};
#if defined(__clang_major__) && __clang_major__ >= 23
    i32x8 gz8 = {0, 0, 0, 0, 0, 0, 0, 0};
    __builtin_amdgcn_tensor_load_to_lds(g0, g1, gz, gz, gz8, 0);
#else
    __builtin_amdgcn_tensor_load_to_lds(g0, g1, gz, gz, 0);
#endif
}
#endif // HAS_TDM

// --------------------------- fp32 -> bf16 convert --------------------------
__global__ void cvt_f32_bf16(const float* __restrict__ src,
                             __bf16* __restrict__ dst, size_t n) {
    size_t i = (size_t)blockIdx.x * blockDim.x + threadIdx.x;
    if (i < n) dst[i] = (__bf16)src[i];
}

// ------------------------------ h0 initialise ------------------------------
__global__ void h_init(const float* __restrict__ h0l,
                       float* __restrict__ hf,      // [2 dirs][2 bufs][B*H]
                       __bf16* __restrict__ hb) {
    int i = blockIdx.x * blockDim.x + threadIdx.x;   // 0 .. 2*B*H-1
    int dir = i >> 15;                               // B*H = 32768
    int off = i & 32767;
    float v = h0l[i];
    hf[dir * 2 * 32768 + off] = v;
    hb[dir * 2 * 32768 + off] = (__bf16)v;
}

// --------------------- gx = x @ W_ih^T + b_ih  (both dirs) -----------------
// xin: [M][K] bf16 row-major.  wih: [2][768][K] bf16.  gx: [2][M][768] f32.
// Block tile 128(M) x 64(N), 8 waves in a 4x2 wave grid, 2x2 WMMA per wave.
// Double-buffered TDM pipeline: tile k+1 is issued before waiting on tile k;
// in-order TDM completion makes s_wait_tensorcnt(2) equivalent to "tile k
// landed", so the DMA runs fully under the WMMA math.
__global__ void __launch_bounds__(256)
gru_gemm_gx(const __bf16* __restrict__ xin, int K,
            const __bf16* __restrict__ wih,
            const float* __restrict__ bih,
            float* __restrict__ gx) {
    __shared__ __align__(16) __bf16 sA[2][128 * 32];   // 2 x 8 KB
    __shared__ __align__(16) __bf16 sB[2][64 * 32];    // 2 x 4 KB

    const int dir = blockIdx.z;
    const __bf16* W = wih + (size_t)dir * G3 * K;
    float* gxd = gx + (size_t)dir * Mrows * G3;

    const int m0 = blockIdx.x * 128;
    const int n0 = blockIdx.y * 64;
    const int tid = threadIdx.x;
    const int lane = tid & 31;
    const int wv = tid >> 5;
    const int wm = (wv >> 1) * 32;
    const int wn = (wv & 1) * 32;
    const int lm = lane & 15;
    const int ksel = (lane >> 4) * 8;
    const int nk = K >> 5;                   // K/32 k-chunks

    v8f acc[2][2] = {};

#if HAS_TDM
    const unsigned sA_addr = (unsigned)(uintptr_t)&sA[0][0];
    const unsigned sB_addr = (unsigned)(uintptr_t)&sB[0][0];

    // Prologue: tile 0 in flight before any compute.
    if (wv == 0) {
        tdm_load_2d(sA_addr, xin + (size_t)m0 * K, 32, 128, (unsigned)K, 1);
        tdm_load_2d(sB_addr, W   + (size_t)n0 * K, 32,  64, (unsigned)K, 1);
    }

    for (int k = 0; k < nk; ++k) {
        const int buf = k & 1;
        // Issue tile k+1 into the other buffer (free since the end-of-iter
        // barrier of k-1), then wait until only its 2 descriptors remain.
        if (wv == 0 && k + 1 < nk) {
            const int k1 = (k + 1) << 5;
            tdm_load_2d(sA_addr + (unsigned)(buf ^ 1) * (128 * 32 * 2),
                        xin + (size_t)m0 * K + k1, 32, 128, (unsigned)K, 1);
            tdm_load_2d(sB_addr + (unsigned)(buf ^ 1) * (64 * 32 * 2),
                        W + (size_t)n0 * K + k1, 32, 64, (unsigned)K, 1);
        }
        if (k + 1 < nk) __builtin_amdgcn_s_wait_tensorcnt(2);
        else            __builtin_amdgcn_s_wait_tensorcnt(0);
        __syncthreads();                     // tile k visible to all waves

#pragma unroll
        for (int mi = 0; mi < 2; ++mi) {
            const __bf16* ap = &sA[buf][(wm + mi * 16 + lm) * 32 + ksel];
            v16bf a = cat16(*(const v8bf*)ap, *(const v8bf*)(ap + 16));
#pragma unroll
            for (int ni = 0; ni < 2; ++ni) {
                const __bf16* bp = &sB[buf][(wn + ni * 16 + lm) * 32 + ksel];
                v16bf b = cat16(*(const v8bf*)bp, *(const v8bf*)(bp + 16));
                acc[mi][ni] = wmma_bf16(a, b, acc[mi][ni]);
            }
        }
        __syncthreads();                     // buffer free for tile k+2
    }
#else
    for (int k0 = 0; k0 < K; k0 += 32) {
        for (int i = tid; i < 128 * 32 / 8; i += 256) {
            int e = i * 8, r = e >> 5, c = e & 31;
            *(v8bf*)&sA[0][r * 32 + c] =
                *(const v8bf*)(xin + (size_t)(m0 + r) * K + k0 + c);
        }
        for (int i = tid; i < 64 * 32 / 8; i += 256) {
            int e = i * 8, r = e >> 5, c = e & 31;
            *(v8bf*)&sB[0][r * 32 + c] =
                *(const v8bf*)(W + (size_t)(n0 + r) * K + k0 + c);
        }
        __syncthreads();
#pragma unroll
        for (int mi = 0; mi < 2; ++mi) {
            const __bf16* ap = &sA[0][(wm + mi * 16 + lm) * 32 + ksel];
            v16bf a = cat16(*(const v8bf*)ap, *(const v8bf*)(ap + 16));
#pragma unroll
            for (int ni = 0; ni < 2; ++ni) {
                const __bf16* bp = &sB[0][(wn + ni * 16 + lm) * 32 + ksel];
                v16bf b = cat16(*(const v8bf*)bp, *(const v8bf*)(bp + 16));
                acc[mi][ni] = wmma_bf16(a, b, acc[mi][ni]);
            }
        }
        __syncthreads();
    }
#endif

#pragma unroll
    for (int mi = 0; mi < 2; ++mi)
#pragma unroll
        for (int ni = 0; ni < 2; ++ni) {
            int col = n0 + wn + ni * 16 + lm;
            float bias = bih[dir * G3 + col];
#pragma unroll
            for (int i = 0; i < 8; ++i) {
                int row = m0 + wm + mi * 16 + i + ((lane >> 4) << 3);
                gxd[(size_t)row * G3 + col] = acc[mi][ni][i] + bias;
            }
        }
}

// ------------------------------- recurrence --------------------------------
// Grid: (8, 2). blockIdx.y = direction; the 8 x-blocks form one workgroup
// cluster per direction. Each WG owns h-columns [j0, j0+32) and computes
// gate columns {j0, j0+256, j0+512}. LDS (dynamic, 112 KB): W_hh slice
// (3 x 32 x 256 bf16 = 48 KB, loaded once by TDM, resident for all 512 steps)
// + hidden state tile (128 x 256 bf16 = 64 KB, one TDM descriptor per step).
// The per-step h load cannot be pipelined: it depends on the previous step's
// stores through the cluster barrier (true recurrence dependence).
__global__ void __launch_bounds__(256)
gru_scan(const float* __restrict__ gx,      // [2][T*B][768] f32
         const __bf16* __restrict__ whhbf,  // [2][768][256] bf16 (this layer)
         const float* __restrict__ bhh,     // [2][768] f32
         float* __restrict__ hf,            // [2][2][B*H] f32 ping-pong
         __bf16* __restrict__ hb,           // [2][2][B*H] bf16 ping-pong
         __bf16* __restrict__ xnext,        // [T*B][512] bf16 next-layer input
         float* __restrict__ outf) {        // d_out (final layer) or nullptr
    extern __shared__ __align__(16) char smem[];
    __bf16* sW = (__bf16*)smem;                      // [3][32][256] = 48 KB
#if HAS_TDM
    __bf16* sH = (__bf16*)(smem + 3 * 32 * 256 * 2); // [128][256] = 64 KB
#endif

    const int dir = blockIdx.y;
    const int j0 = blockIdx.x * 32;
    const int tid = threadIdx.x;
    const int lane = tid & 31;
    const int wv = tid >> 5;
    const int mw = wv * 16;
    const int lm = lane & 15;
    const int ksel = (lane >> 4) * 8;

    const __bf16* Wd = whhbf + (size_t)dir * G3 * Hh;

    // Stage W_hh slice once: sW[gate][rr][k] with rr = column-j0 offset (0..31)
#if HAS_TDM
    if (wv == 0) {
        unsigned sW_addr = (unsigned)(uintptr_t)&sW[0];
#pragma unroll
        for (int gate = 0; gate < 3; ++gate)
            tdm_load_2d(sW_addr + (unsigned)gate * 32 * 256 * 2,
                        Wd + (size_t)(gate * 256 + j0) * Hh, 256, 32, 256, 1);
    }
    __builtin_amdgcn_s_wait_tensorcnt(0);
#else
    for (int i = tid; i < 3 * 32 * 256 / 8; i += 256) {
        int e = i * 8;
        int row = e >> 8;                    // 0..95 = gate*32 + rr
        int k = e & 255;
        int gate = row >> 5, rr = row & 31;
        *(v8bf*)&sW[e] =
            *(const v8bf*)(Wd + (size_t)(gate * 256 + j0 + rr) * Hh + k);
    }
#endif
    __syncthreads();

    const float* gxd = gx + (size_t)dir * Mrows * G3;
    const float* bh = bhh + dir * G3;
#if HAS_TDM
    const unsigned sH_addr = (unsigned)(uintptr_t)&sH[0];
#endif

    for (int s = 0; s < Tt; ++s) {
        const int t = dir ? (Tt - 1 - s) : s;
        const float*  hcf = hf + dir * 2 * 32768 + (s & 1) * 32768;
        float*        hnf = hf + dir * 2 * 32768 + ((s + 1) & 1) * 32768;
        const __bf16* hcb = hb + dir * 2 * 32768 + (s & 1) * 32768;
        __bf16*       hnb = hb + dir * 2 * 32768 + ((s + 1) & 1) * 32768;

#if HAS_TDM
        // DMA the full hidden state (128x256 bf16) into LDS for this step.
        if (wv == 0) tdm_load_2d(sH_addr, hcb, 256, 128, 256, 1);
        __builtin_amdgcn_s_wait_tensorcnt(0);
        __syncthreads();
        const __bf16* hA = sH;
#else
        const __bf16* hA = hcb;
#endif

        // gh = h @ W_hh^T for this WG's 96 gate columns
        v8f acc[6] = {};
#pragma unroll
        for (int kc = 0; kc < 8; ++kc) {
            const __bf16* ap = hA + (size_t)(mw + lm) * Hh + kc * 32 + ksel;
            v16bf a = cat16(*(const v8bf*)ap, *(const v8bf*)(ap + 16));
#pragma unroll
            for (int n = 0; n < 6; ++n) {
                const __bf16* bp =
                    &sW[((n >> 1) * 32 + (n & 1) * 16 + lm) * 256 + kc * 32 + ksel];
                v16bf b = cat16(*(const v8bf*)bp, *(const v8bf*)(bp + 16));
                acc[n] = wmma_bf16(a, b, acc[n]);
            }
        }

        const float* gxt = gxd + (size_t)t * Bb * G3;

        // Prefetch next step's gx stream (global_prefetch_b8)
        if (s + 1 < Tt) {
            int tn = dir ? (Tt - 2 - s) : (s + 1);
            __builtin_prefetch(gxd + (size_t)tn * Bb * G3 + (size_t)mw * G3 + j0, 0, 1);
        }

        // Gates + state update for owned columns
#pragma unroll
        for (int u = 0; u < 2; ++u) {
            int col = j0 + u * 16 + lm;
#pragma unroll
            for (int i = 0; i < 8; ++i) {
                int row = mw + i + ((lane >> 4) << 3);
                const float* gr = gxt + (size_t)row * G3 + col;
                float r = sigmoidf(gr[0]   + acc[0 + u][i] + bh[col]);
                float z = sigmoidf(gr[256] + acc[2 + u][i] + bh[col + 256]);
                float n = tanhf(gr[512] + r * (acc[4 + u][i] + bh[col + 512]));
                float hp = hcf[row * Hh + col];
                float hv = (1.0f - z) * n + z * hp;
                hnf[row * Hh + col] = hv;
                hnb[row * Hh + col] = (__bf16)hv;
                size_t ob = ((size_t)t * Bb + row) * 512 + dir * 256 + col;
                xnext[ob] = (__bf16)hv;
                if (outf) outf[ob] = hv;
            }
        }

        // Make h_next visible across the cluster, then step-sync all 8 WGPs.
        __threadfence();
        __syncthreads();
        __builtin_amdgcn_s_cluster_barrier();   // s_barrier_signal/wait -3
    }
}

// ------------------------------- host side ---------------------------------
extern "C" void kernel_launch(void* const* d_in, const int* in_sizes, int n_in,
                              void* d_out, int out_size, void* d_ws, size_t ws_size,
                              hipStream_t stream) {
    const float* x       = (const float*)d_in[0];   // [T][B][128]
    const float* h0      = (const float*)d_in[1];   // [12][B][H]
    const float* w_ih_l0 = (const float*)d_in[2];   // [2][768][128]
    const float* w_hh_l0 = (const float*)d_in[3];   // [2][768][256]
    const float* b_ih_l0 = (const float*)d_in[4];   // [2][768]
    const float* b_hh_l0 = (const float*)d_in[5];   // [2][768]
    const float* w_ih    = (const float*)d_in[6];   // [5][2][768][512]
    const float* w_hh    = (const float*)d_in[7];   // [5][2][768][256]
    const float* b_ih    = (const float*)d_in[8];   // [5][2][768]
    const float* b_hh    = (const float*)d_in[9];   // [5][2][768]
    float* out = (float*)d_out;                     // [T][B][512]

    // Workspace carve-up
    char* p = (char*)d_ws;
    __bf16* xA = (__bf16*)p;  p += (size_t)Mrows * 512 * sizeof(__bf16);
    __bf16* xB = (__bf16*)p;  p += (size_t)Mrows * 512 * sizeof(__bf16);
    float*  gx = (float*)p;   p += (size_t)2 * Mrows * G3 * sizeof(float);
    __bf16* wihb = (__bf16*)p;
    const size_t wih_l0_n = (size_t)2 * G3 * 128;
    const size_t wih_hi_n = (size_t)5 * 2 * G3 * 512;
    p += (wih_l0_n + wih_hi_n) * sizeof(__bf16);
    __bf16* whhb = (__bf16*)p; p += (size_t)6 * 2 * G3 * Hh * sizeof(__bf16);
    float*  hfb = (float*)p;   p += (size_t)2 * 2 * Bb * Hh * sizeof(float);
    __bf16* hbb = (__bf16*)p;  p += (size_t)2 * 2 * Bb * Hh * sizeof(__bf16);

    auto cvt = [&](const float* s, __bf16* d, size_t n) {
        int blocks = (int)((n + 255) / 256);
        cvt_f32_bf16<<<blocks, 256, 0, stream>>>(s, d, n);
    };

    // One-time weight / input conversions to bf16
    cvt(w_ih_l0, wihb, wih_l0_n);
    cvt(w_ih,    wihb + wih_l0_n, wih_hi_n);
    cvt(w_hh_l0, whhb, (size_t)2 * G3 * Hh);
    cvt(w_hh,    whhb + (size_t)2 * G3 * Hh, (size_t)5 * 2 * G3 * Hh);
    cvt(x, xA, (size_t)Mrows * 128);

    const int scan_lds = (3 * 32 * 256 + 128 * 256) * (int)sizeof(__bf16); // 112 KB

    __bf16* cur = xA;
    __bf16* nxt = xB;
    for (int l = 0; l < 6; ++l) {
        const int K = l ? 512 : 128;
        const __bf16* wl = l ? (wihb + wih_l0_n + (size_t)(l - 1) * 2 * G3 * 512)
                             : wihb;
        const float* bi = l ? (b_ih + (size_t)(l - 1) * 2 * G3) : b_ih_l0;
        const float* bhp = l ? (b_hh + (size_t)(l - 1) * 2 * G3) : b_hh_l0;
        const __bf16* wh = whhb + (size_t)l * 2 * G3 * Hh;

        gru_gemm_gx<<<dim3(Mrows / 128, G3 / 64, 2), 256, 0, stream>>>(
            cur, K, wl, bi, gx);

        h_init<<<(2 * Bb * Hh) / 256, 256, 0, stream>>>(
            h0 + (size_t)2 * l * Bb * Hh, hfb, hbb);

        gru_scan<<<dim3(8, 2), 256, scan_lds, stream>>>(
            gx, wh, bhp, hfb, hbb, nxt, (l == 5) ? out : nullptr);

        __bf16* tmp = cur; cur = nxt; nxt = tmp;
    }
}
